// displacement_decoder_5995774345566
// MI455X (gfx1250) — matrix operand order repaired and checked by reading, not compiled, not run on hardware
//
#include <hip/hip_runtime.h>
#include <hip/hip_bf16.h>
#include <hip/hip_fp16.h>

typedef __attribute__((ext_vector_type(16))) _Float16 v16h;
typedef __attribute__((ext_vector_type(8)))  _Float16 v8h;
typedef __attribute__((ext_vector_type(8)))  float    v8f;

#define NPTS       50000
#define NB         8
#define HID        128
#define LDS_STRIDE 136          // halves per staged row (272B => rows on distinct banks)
#define WAVES      4            // waves per block
#define ROWS       32           // rows (points) per wave = 2 M-tiles sharing B fragments

// d_ws layout in halves
#define W0_OFF     0            // [128][96]   lin0, K padded 92->96
#define WD_OFF     12288        // [6][128][128]
#define W1A_OFF    110592       // [16][128]   lin1a, M padded 3->16
#define WS_HALVES  112640

// ---------------- prep: fp32 weights -> fp16 with WMMA-friendly padding ----------------
__global__ void prep_weights_kernel(const float* __restrict__ w_lin0,
                                    const float* __restrict__ w_deform,
                                    const float* __restrict__ w_lin1a,
                                    _Float16* __restrict__ wsh) {
    int idx = blockIdx.x * blockDim.x + threadIdx.x;
    if (idx >= WS_HALVES) return;
    float v;
    if (idx < WD_OFF) {
        int o = idx / 96, k = idx - o * 96;
        v = (k < 92) ? w_lin0[o * 92 + k] : 0.0f;
    } else if (idx < W1A_OFF) {
        v = w_deform[idx - WD_OFF];
    } else {
        int i3 = idx - W1A_OFF;
        int o = i3 >> 7, k = i3 & 127;
        v = (o < 3) ? w_lin1a[o * 128 + k] : 0.0f;
    }
    wsh[idx] = (_Float16)v;
}

// A fragment 16x32 f16: lane = row, 16 contiguous K halves (matches load_b_frag)
__device__ __forceinline__ v16h load_a_frag(const _Float16* X, int k0, int lane) {
    int m   = lane & 15;
    int sub = (lane >> 4) << 4;              // 0 or 16
    const _Float16* p = X + m * LDS_STRIDE + k0 + sub;
    v8h lo = *(const v8h*)p;                 // ds_load_b128
    v8h hi = *(const v8h*)(p + 8);           // ds_load_b128
    return __builtin_shufflevector(lo, hi, 0,1,2,3,4,5,6,7,8,9,10,11,12,13,14,15);
}

// B fragment 32x16 f16: lane = output column, 16 contiguous K halves (32B load)
__device__ __forceinline__ v16h load_b_frag(const _Float16* __restrict__ W, int ldw,
                                            int n0, int k0, int lane) {
    int n  = n0 + (lane & 15);
    int kk = k0 + ((lane >> 4) << 4);
    return *(const v16h*)(W + (size_t)n * ldw + kk);
}

// C/D 16x16 f32 -> f16 A-rows in LDS (optional ELU)
__device__ __forceinline__ void restage(_Float16* X, const v8f* acc, int lane, bool act) {
    int col = lane & 15;
    int rhi = (lane >> 4) << 3;
    #pragma unroll
    for (int t = 0; t < 8; ++t) {
        #pragma unroll
        for (int i = 0; i < 8; ++i) {
            float v = acc[t][i];
            if (act) v = v > 0.0f ? v : (__expf(v) - 1.0f);
            X[(i + rhi) * LDS_STRIDE + t * 16 + col] = (_Float16)v;
        }
    }
}

// ---------------- fused MLP: 1 wave == 32 points (two M-tiles share weights) ----------------
__global__ __launch_bounds__(WAVES * 32) void displacement_decoder_kernel(
    const float* __restrict__ z0, const float* __restrict__ cons,
    const _Float16* __restrict__ wsh, const float* __restrict__ w1b,
    float* __restrict__ out, int ntiles)
{
    __shared__ _Float16 smem[WAVES * ROWS * LDS_STRIDE];
    const int lane = threadIdx.x & 31;
    const int wave = threadIdx.x >> 5;
    _Float16* X0 = &smem[wave * ROWS * LDS_STRIDE];       // rows 0..15
    _Float16* X1 = X0 + 16 * LDS_STRIDE;                  // rows 16..31

    const int tile = blockIdx.x * WAVES + wave;
    if (tile >= ntiles) return;                           // wave-uniform
    const int g0 = tile * ROWS;

    // ---- stage input features [32][96]: 60 posenc + 32 z0 + 4 zero pad ----
    for (int r = 0; r < ROWS; ++r) {
        int g = g0 + r;
        int b = g / NPTS, n = g - b * NPTS;
        float c3[3];
        c3[0] = cons[n * 3 + 0];
        c3[1] = cons[n * 3 + 1];
        c3[2] = cons[n * 3 + 2];
        #pragma unroll
        for (int p = 0; p < 3; ++p) {
            int f = p * 32 + lane;
            float val;
            if (f < 60) {
                int c = f / 20, j = f - c * 20;
                int e = (j < 10) ? j : j - 10;
                // freq = 128 * (1/128)^(e/9) = exp(ln128 * (1 - e/9))
                float freq = __expf(4.852030263919617f * (1.0f - e * (1.0f / 9.0f)));
                float k = c3[c] * freq;
                val = (j < 10) ? __sinf(k) : __cosf(k);
            } else if (f < 92) {
                val = z0[b * 32 + (f - 60)];
            } else {
                val = 0.0f;
            }
            X0[r * LDS_STRIDE + f] = (_Float16)val;
        }
    }

    v16h a0[4], a1[4];
    v8f  acc0[8], acc1[8];

    // ---- lin0: K=96 (3 chunks), no activation ----
    const _Float16* W0 = wsh + W0_OFF;
    #pragma unroll
    for (int kc = 0; kc < 3; ++kc) { a0[kc] = load_a_frag(X0, kc * 32, lane);
                                     a1[kc] = load_a_frag(X1, kc * 32, lane); }
    #pragma unroll
    for (int t = 0; t < 8; ++t) {
        v8f c0 = {}, c1 = {};
        #pragma unroll
        for (int kc = 0; kc < 3; ++kc) {
            v16h b = load_b_frag(W0, 96, t * 16, kc * 32, lane);
            c0 = __builtin_amdgcn_wmma_f32_16x16x32_f16(false, a0[kc], false, b,
                                                        (short)0, c0, false, false);
            c1 = __builtin_amdgcn_wmma_f32_16x16x32_f16(false, a1[kc], false, b,
                                                        (short)0, c1, false, false);
        }
        acc0[t] = c0; acc1[t] = c1;
    }
    restage(X0, acc0, lane, false);
    restage(X1, acc1, lane, false);

    // ---- 6x ELU(128x128): each B fragment feeds two WMMAs ----
    for (int layer = 0; layer < 6; ++layer) {
        const _Float16* Wd = wsh + WD_OFF + layer * HID * HID;
        const _Float16* Wn = (layer < 5) ? (Wd + HID * HID) : (wsh + W1A_OFF);
        __builtin_prefetch(Wn, 0, 0);                     // global_prefetch_b8
        #pragma unroll
        for (int kc = 0; kc < 4; ++kc) { a0[kc] = load_a_frag(X0, kc * 32, lane);
                                         a1[kc] = load_a_frag(X1, kc * 32, lane); }
        #pragma unroll
        for (int t = 0; t < 8; ++t) {
            v8f c0 = {}, c1 = {};
            #pragma unroll
            for (int kc = 0; kc < 4; ++kc) {
                v16h b = load_b_frag(Wd, HID, t * 16, kc * 32, lane);
                c0 = __builtin_amdgcn_wmma_f32_16x16x32_f16(false, a0[kc], false, b,
                                                            (short)0, c0, false, false);
                c1 = __builtin_amdgcn_wmma_f32_16x16x32_f16(false, a1[kc], false, b,
                                                            (short)0, c1, false, false);
            }
            acc0[t] = c0; acc1[t] = c1;
        }
        restage(X0, acc0, lane, true);
        restage(X1, acc1, lane, true);
    }

    // ---- lin1a (128 -> 3, padded to 16): one N-tile, both M-tiles ----
    const _Float16* W1 = wsh + W1A_OFF;
    #pragma unroll
    for (int kc = 0; kc < 4; ++kc) { a0[kc] = load_a_frag(X0, kc * 32, lane);
                                     a1[kc] = load_a_frag(X1, kc * 32, lane); }
    v8f r0acc = {}, r1acc = {};
    #pragma unroll
    for (int kc = 0; kc < 4; ++kc) {
        v16h b = load_b_frag(W1, HID, 0, kc * 32, lane);
        r0acc = __builtin_amdgcn_wmma_f32_16x16x32_f16(false, a0[kc], false, b,
                                                       (short)0, r0acc, false, false);
        r1acc = __builtin_amdgcn_wmma_f32_16x16x32_f16(false, a1[kc], false, b,
                                                       (short)0, r1acc, false, false);
    }

    // ---- ELU, park cols 0..2 of all 32 rows as f32 in LDS ----
    float* resb = (float*)X0;                             // 32 rows x 4 floats = 512B
    {
        int col = lane & 15;
        int rhi = (lane >> 4) << 3;
        if (col < 3) {
            #pragma unroll
            for (int i = 0; i < 8; ++i) {
                float v = r0acc[i];
                v = v > 0.0f ? v : (__expf(v) - 1.0f);
                resb[(i + rhi) * 4 + col] = v;
                float w = r1acc[i];
                w = w > 0.0f ? w : (__expf(w) - 1.0f);
                resb[(16 + i + rhi) * 4 + col] = w;
            }
        }
    }

    // ---- lin1b (3x3) + cons add; each lane = one of the 32 points ----
    {
        int g = g0 + lane;
        int b = g / NPTS, n = g - b * NPTS;
        (void)b;
        float r0 = resb[lane * 4 + 0];
        float r1 = resb[lane * 4 + 1];
        float r2 = resb[lane * 4 + 2];
        const size_t half = (size_t)NB * NPTS * 3;
        #pragma unroll
        for (int o = 0; o < 3; ++o) {
            float rv = w1b[o * 3 + 0] * r0 + w1b[o * 3 + 1] * r1 + w1b[o * 3 + 2] * r2;
            float pv = cons[n * 3 + o] + rv;
            out[(size_t)g * 3 + o]        = pv;   // pos
            out[half + (size_t)g * 3 + o] = rv;   // resn
        }
    }
}

extern "C" void kernel_launch(void* const* d_in, const int* in_sizes, int n_in,
                              void* d_out, int out_size, void* d_ws, size_t ws_size,
                              hipStream_t stream) {
    const float* z0       = (const float*)d_in[0];
    const float* cons     = (const float*)d_in[1];
    const float* w_lin0   = (const float*)d_in[2];
    const float* w_deform = (const float*)d_in[3];
    const float* w_lin1a  = (const float*)d_in[4];
    const float* w_lin1b  = (const float*)d_in[5];
    float* out = (float*)d_out;
    _Float16* wsh = (_Float16*)d_ws;                 // uses 225,280 B of scratch

    prep_weights_kernel<<<(WS_HALVES + 255) / 256, 256, 0, stream>>>(
        w_lin0, w_deform, w_lin1a, wsh);

    const int ntiles = (NB * NPTS) / ROWS;           // 12,500 tiles of 32 points
    const int blocks = (ntiles + WAVES - 1) / WAVES; // 3,125 blocks x 128 threads
    displacement_decoder_kernel<<<blocks, WAVES * 32, 0, stream>>>(
        z0, cons, wsh, w_lin1b, out, ntiles);
}
